// GPT_33028298506745
// MI455X (gfx1250) — compile-verified, matrix operand order checked by reading, not run on hardware
//
#include <hip/hip_runtime.h>
#include <stdint.h>

// GPT forward for MI455X (gfx1250, wave32, WMMA, async-LDS staging).
// B=2 T=1024 C=512 H=8 D=64 L=4 V=32000
#define B_  2
#define T_  1024
#define C_  512
#define H_  8
#define D_  64
#define L_  4
#define V_  32000
#define FF_ 2048
#define BT_ (B_*T_)

typedef __attribute__((ext_vector_type(16))) __bf16 v16bf;
typedef __attribute__((ext_vector_type(8)))  float  v8f;

union FragU { v16bf v; uint4 u[2]; };

__device__ inline uint16_t f2bf(float f) {
  uint32_t u = __float_as_uint(f);
  u += 0x7fffu + ((u >> 16) & 1u);       // round-to-nearest-even
  return (uint16_t)(u >> 16);
}

__device__ inline v8f vzero() {
  v8f z;
  for (int i = 0; i < 8; i++) z[i] = 0.0f;
  return z;
}

// Load a 16x32 bf16 A/B fragment: two 16B chunks (K 0..7+half, K 16..23+half).
__device__ inline v16bf load_frag(const uint16_t* p0, const uint16_t* p1) {
  FragU f;
  f.u[0] = *reinterpret_cast<const uint4*>(p0);
  f.u[1] = *reinterpret_cast<const uint4*>(p1);
  return f.v;
}

__device__ inline v8f wmma_bf16(v16bf a, v16bf b, v8f c) {
  return __builtin_amdgcn_wmma_f32_16x16x32_bf16(false, a, false, b,
                                                 (short)0, c, false, false);
}

// Async copy 16B global -> LDS (ASYNCcnt-tracked; lane-private 16B).
// LDS byte address = low 32 bits of the generic shared pointer.
__device__ inline void async_cp16(uint32_t ldsaddr, const void* gaddr) {
  asm volatile("global_load_async_to_lds_b128 %0, %1, off"
               :: "v"(ldsaddr), "v"((uint64_t)(uintptr_t)gaddr)
               : "memory");
}

__device__ inline void wait_async0() {
  asm volatile("s_wait_asynccnt 0x0" ::: "memory");
}

// ---------------------------------------------------------------------------
// Weight prep: f32 [K][N] -> bf16 [N][K] (transposed so B-fragments are
// contiguous 16B loads per lane).
// ---------------------------------------------------------------------------
__global__ void convT_bf16(const float* __restrict__ W,
                           uint16_t* __restrict__ WT, int K, int N) {
  size_t i = (size_t)blockIdx.x * blockDim.x + threadIdx.x;
  size_t total = (size_t)K * N;
  if (i >= total) return;
  size_t kk = i / N, n = i % N;
  WT[n * (size_t)K + kk] = f2bf(W[i]);
}

// ---------------------------------------------------------------------------
// Embedding: x[b,t,:] = tok_emb[idx[b,t],:] + pos_emb[t,:]
// ---------------------------------------------------------------------------
__global__ void embed_kernel(const int* __restrict__ idx,
                             const float* __restrict__ tok,
                             const float* __restrict__ pos,
                             float* __restrict__ x) {
  int row = blockIdx.x;                 // b*T + t
  int t = row % T_;
  int token = idx[row];
  for (int c = threadIdx.x; c < C_; c += blockDim.x)
    x[(size_t)row * C_ + c] = tok[(size_t)token * C_ + c] + pos[(size_t)t * C_ + c];
}

// ---------------------------------------------------------------------------
// LayerNorm, f32 in -> bf16 out. One block of 256 threads per row (C=512).
// ---------------------------------------------------------------------------
__global__ __launch_bounds__(256) void layernorm_bf16(
    const float* __restrict__ x, const float* __restrict__ w,
    const float* __restrict__ bvec, uint16_t* __restrict__ out) {
  __shared__ float red[256];
  int row = blockIdx.x;
  int tid = threadIdx.x;
  const float* xr = x + (size_t)row * C_;
  float a = xr[tid], b = xr[tid + 256];
  red[tid] = a + b;
  __syncthreads();
  for (int off = 128; off > 0; off >>= 1) {
    if (tid < off) red[tid] += red[tid + off];
    __syncthreads();
  }
  float mean = red[0] * (1.0f / C_);
  __syncthreads();
  float d1 = a - mean, d2 = b - mean;
  red[tid] = d1 * d1 + d2 * d2;
  __syncthreads();
  for (int off = 128; off > 0; off >>= 1) {
    if (tid < off) red[tid] += red[tid + off];
    __syncthreads();
  }
  float rstd = rsqrtf(red[0] * (1.0f / C_) + 1e-5f);
  out[(size_t)row * C_ + tid]       = f2bf(d1 * rstd * w[tid]       + bvec[tid]);
  out[(size_t)row * C_ + tid + 256] = f2bf(d2 * rstd * w[tid + 256] + bvec[tid + 256]);
}

// ---------------------------------------------------------------------------
// WMMA GEMM with double-buffered async-LDS staging.
// out[M,N] = epilogue( A[M,K](bf16) @ BT[N,K](bf16)^T + bias )
// 256 threads = 8 waves; block tile 128x128; wave tile 32x64 (2x4 wmma).
// Per k-step: stage next 128x32 A and B tiles via global_load_async_to_lds_b128
// while WMMAs consume the current LDS tiles; s_wait_asynccnt + barrier flips.
// All dims are multiples of the tile sizes (M=2048, N in {512,2048,32000},
// K in {512,2048}) -> no bounds checks.
// ---------------------------------------------------------------------------
__global__ __launch_bounds__(256) void gemm_bf16(
    const uint16_t* __restrict__ A, const uint16_t* __restrict__ BT,
    const float* __restrict__ bias, const float* __restrict__ resid,
    float* __restrict__ outF, uint16_t* __restrict__ outB,
    int M, int N, int K, int gelu) {
  (void)M;
  __shared__ __align__(16) uint16_t sA[2][128][32];   // 8KB x2
  __shared__ __align__(16) uint16_t sB[2][128][32];   // 8KB x2

  const int tid   = threadIdx.x;
  const int lane  = tid & 31;
  const int wave  = tid >> 5;
  const int l16   = lane & 15;
  const int khalf = (lane >> 4) * 8;        // K offset (halfs) AND row offset
  const int wm = (wave & 3) * 32;
  const int wn = (wave >> 2) * 64;
  const int bm0 = blockIdx.y * 128;
  const int bn0 = blockIdx.x * 128;

  v8f acc[2][4];
  for (int mi = 0; mi < 2; mi++)
    for (int ni = 0; ni < 4; ni++) acc[mi][ni] = vzero();

  // Each thread stages 2 x 16B chunks of A and of B per k-step.
  const int c0 = tid * 2;
  auto stage = [&](int buf, int k0) {
    for (int t = 0; t < 2; t++) {
      int c   = c0 + t;
      int row = c >> 2;             // 0..127
      int ko  = (c & 3) * 8;        // 0,8,16,24 halfs
      async_cp16((uint32_t)(uintptr_t)&sA[buf][row][ko],
                 A + (size_t)(bm0 + row) * K + k0 + ko);
      async_cp16((uint32_t)(uintptr_t)&sB[buf][row][ko],
                 BT + (size_t)(bn0 + row) * K + k0 + ko);
    }
  };

  stage(0, 0);
  wait_async0();
  __syncthreads();

  const int nk = K / 32;
  for (int ks = 0; ks < nk; ks++) {
    const int cur = ks & 1;
    if (ks + 1 < nk) stage(cur ^ 1, (ks + 1) * 32);   // overlap with compute

    v16bf af[2], bfr[4];
    for (int mi = 0; mi < 2; mi++) {
      const uint16_t* p = &sA[cur][wm + mi * 16 + l16][khalf];
      af[mi] = load_frag(p, p + 16);
    }
    for (int ni = 0; ni < 4; ni++) {
      const uint16_t* p = &sB[cur][wn + ni * 16 + l16][khalf];
      bfr[ni] = load_frag(p, p + 16);
    }
    for (int mi = 0; mi < 2; mi++)
      for (int ni = 0; ni < 4; ni++)
        acc[mi][ni] = wmma_bf16(af[mi], bfr[ni], acc[mi][ni]);

    wait_async0();
    __syncthreads();
  }

  for (int mi = 0; mi < 2; mi++)
    for (int ni = 0; ni < 4; ni++)
      for (int r = 0; r < 8; r++) {
        int row = bm0 + wm + mi * 16 + khalf + r;  // lanes16-31 -> +8 rows
        int col = bn0 + wn + ni * 16 + l16;
        float v = acc[mi][ni][r];
        if (bias) v += bias[col];
        if (gelu) v = 0.5f * v * (1.0f + erff(v * 0.70710678118f));
        size_t o = (size_t)row * N + col;
        if (outB) {
          outB[o] = f2bf(v);
        } else {
          if (resid) v += resid[o];
          outF[o] = v;
        }
      }
}

// ---------------------------------------------------------------------------
// Flash attention (causal). One block per (b,h) x 128 query rows; 8 waves,
// each wave owns 16 query rows. Key tiles of 32. K tile async-staged to LDS
// (shared by all 8 waves), V tile staged transposed (manual), S = Q K^T via
// WMMA, online softmax with width-16 shuffles, P re-laid-out via LDS into an
// A-fragment, P@V via WMMA from LDS.
// ---------------------------------------------------------------------------
__global__ __launch_bounds__(256) void attn_flash(
    const uint16_t* __restrict__ q, const uint16_t* __restrict__ k,
    const uint16_t* __restrict__ v, uint16_t* __restrict__ y) {
  __shared__ __align__(16) uint16_t sK[32][64];         // [key][d]   4KB
  __shared__ __align__(16) uint16_t ldsVT[D_][32];      // [d][key]   4KB
  __shared__ __align__(16) uint16_t ldsP[8][16][32];    // per-wave P 8KB

  const int tid   = threadIdx.x;
  const int lane  = tid & 31;
  const int wave  = tid >> 5;
  const int l16   = lane & 15;
  const int khalf = (lane >> 4) * 8;

  const int bh = blockIdx.y;
  const int b  = bh / H_;
  const int h  = bh % H_;
  const int q0 = blockIdx.x * 128;
  const int m0 = q0 + wave * 16;
  const size_t rowbase = (size_t)b * T_;
  const int hoff = h * D_;
  const float scale = 0.125f;           // 1/sqrt(64)

  // Q fragments for this wave's 16 rows, two K-chunks of 32 over D=64
  v16bf qf[2];
  for (int c = 0; c < 2; c++) {
    const uint16_t* p = q + (rowbase + m0 + l16) * C_ + hoff + c * 32 + khalf;
    qf[c] = load_frag(p, p + 16);
  }

  v8f o[4];
  for (int ni = 0; ni < 4; ni++) o[ni] = vzero();
  float mrow[8], lrow[8];
  for (int r = 0; r < 8; r++) { mrow[r] = -3.0e38f; lrow[r] = 0.0f; }

  for (int s0 = 0; s0 < q0 + 128; s0 += 32) {
    __syncthreads();
    {   // async stage K[s0..s0+31][hoff..hoff+63] -> sK (one 16B chunk/thread)
      int row = tid >> 3;              // key 0..31
      int ko  = (tid & 7) * 8;         // d 0..56
      async_cp16((uint32_t)(uintptr_t)&sK[row][ko],
                 k + (rowbase + s0 + row) * C_ + hoff + ko);
    }
    {   // manual transposed stage: V -> ldsVT[d][key]
      int si = tid >> 3;               // key 0..31
      int dg = (tid & 7) * 8;          // d group 0..56
      const uint16_t* p = v + (rowbase + s0 + si) * C_ + hoff + dg;
      uint4 d0 = *reinterpret_cast<const uint4*>(p);
      const uint16_t* e = reinterpret_cast<const uint16_t*>(&d0);
      for (int j = 0; j < 8; j++) ldsVT[dg + j][si] = e[j];
    }
    wait_async0();
    __syncthreads();

    if (s0 <= m0 + 15) {               // tile intersects causal region
      float p2[2][8];
      for (int sub = 0; sub < 2; sub++) {
        int n0 = s0 + sub * 16;
        v8f s = vzero();
        for (int c = 0; c < 2; c++) {
          const uint16_t* kp = &sK[sub * 16 + l16][c * 32 + khalf];
          v16bf kf = load_frag(kp, kp + 16);
          s = wmma_bf16(qf[c], kf, s);
        }
        for (int r = 0; r < 8; r++) {
          int t = m0 + khalf + r;
          int sidx = n0 + l16;
          float val = s[r] * scale;
          p2[sub][r] = (sidx > t) ? -1.0e30f : val;
        }
      }
      // online softmax per row (rows live across 16 lanes of each half-wave)
      for (int r = 0; r < 8; r++) {
        float mx = fmaxf(p2[0][r], p2[1][r]);
        for (int off = 8; off > 0; off >>= 1)
          mx = fmaxf(mx, __shfl_xor(mx, off, 16));
        float mnew  = fmaxf(mrow[r], mx);
        float alpha = __expf(mrow[r] - mnew);
        for (int ni = 0; ni < 4; ni++) o[ni][r] = o[ni][r] * alpha;
        float e0 = __expf(p2[0][r] - mnew);
        float e1 = __expf(p2[1][r] - mnew);
        p2[0][r] = e0; p2[1][r] = e1;
        float ps = e0 + e1;
        for (int off = 8; off > 0; off >>= 1)
          ps += __shfl_xor(ps, off, 16);
        lrow[r] = lrow[r] * alpha + ps;
        mrow[r] = mnew;
      }
      // P (C/D layout) -> LDS -> A fragment (bf16)
      for (int sub = 0; sub < 2; sub++)
        for (int r = 0; r < 8; r++)
          ldsP[wave][khalf + r][sub * 16 + l16] = f2bf(p2[sub][r]);
      asm volatile("s_wait_dscnt 0x0" ::);
      const uint16_t* pp = &ldsP[wave][l16][khalf];
      v16bf pf = load_frag(pp, pp + 16);
      for (int ni = 0; ni < 4; ni++) {
        const uint16_t* vp = &ldsVT[ni * 16 + l16][khalf];
        v16bf vf = load_frag(vp, vp + 16);
        o[ni] = wmma_bf16(pf, vf, o[ni]);
      }
    }
  }

  for (int r = 0; r < 8; r++) {
    float inv = 1.0f / lrow[r];
    int t = m0 + khalf + r;
    for (int ni = 0; ni < 4; ni++)
      y[(rowbase + t) * C_ + hoff + ni * 16 + l16] = f2bf(o[ni][r] * inv);
  }
}

// ---------------------------------------------------------------------------
// Host-side orchestration
// ---------------------------------------------------------------------------
extern "C" void kernel_launch(void* const* d_in, const int* in_sizes, int n_in,
                              void* d_out, int out_size, void* d_ws, size_t ws_size,
                              hipStream_t stream) {
  (void)in_sizes; (void)n_in; (void)out_size; (void)ws_size;

  const int*   idx  = (const int*)  d_in[0];
  const float* tok  = (const float*)d_in[1];
  const float* pos  = (const float*)d_in[2];
  const float* ln1w = (const float*)d_in[3];
  const float* ln1b = (const float*)d_in[4];
  const float* Wq   = (const float*)d_in[5];
  const float* bq   = (const float*)d_in[6];
  const float* Wk   = (const float*)d_in[7];
  const float* bk   = (const float*)d_in[8];
  const float* Wv   = (const float*)d_in[9];
  const float* bv   = (const float*)d_in[10];
  const float* Wo   = (const float*)d_in[11];
  const float* bo   = (const float*)d_in[12];
  const float* ln2w = (const float*)d_in[13];
  const float* ln2b = (const float*)d_in[14];
  const float* W1   = (const float*)d_in[15];
  const float* b1   = (const float*)d_in[16];
  const float* W2   = (const float*)d_in[17];
  const float* b2   = (const float*)d_in[18];
  const float* lnfw = (const float*)d_in[19];
  const float* lnfb = (const float*)d_in[20];
  const float* Wh   = (const float*)d_in[21];
  float* logits = (float*)d_out;

  // Workspace carve (all 256B aligned). Total ~79 MB.
  char* ws = (char*)d_ws;
  size_t off = 0;
  auto carve = [&](size_t bytes) -> char* {
    char* p = ws + off;
    off += (bytes + 255) & ~(size_t)255;
    return p;
  };
  float*    x   = (float*)   carve(sizeof(float) * BT_ * C_);
  uint16_t* hb  = (uint16_t*)carve(2ull * BT_ * C_);
  uint16_t* qb  = (uint16_t*)carve(2ull * BT_ * C_);
  uint16_t* kb  = (uint16_t*)carve(2ull * BT_ * C_);
  uint16_t* vb  = (uint16_t*)carve(2ull * BT_ * C_);
  uint16_t* yb  = (uint16_t*)carve(2ull * BT_ * C_);
  uint16_t* gb  = (uint16_t*)carve(2ull * BT_ * FF_);
  uint16_t* WqT = (uint16_t*)carve(2ull * L_ * C_ * C_);
  uint16_t* WkT = (uint16_t*)carve(2ull * L_ * C_ * C_);
  uint16_t* WvT = (uint16_t*)carve(2ull * L_ * C_ * C_);
  uint16_t* WoT = (uint16_t*)carve(2ull * L_ * C_ * C_);
  uint16_t* W1T = (uint16_t*)carve(2ull * L_ * C_ * FF_);
  uint16_t* W2T = (uint16_t*)carve(2ull * L_ * FF_ * C_);
  uint16_t* WhT = (uint16_t*)carve(2ull * C_ * V_);

  const int TPB = 256;
  // --- weight prep: bf16 + transpose ---
  for (int i = 0; i < L_; i++) {
    size_t wo = (size_t)i * C_ * C_;
    int nCC = (C_ * C_ + TPB - 1) / TPB;
    convT_bf16<<<nCC, TPB, 0, stream>>>(Wq + wo, WqT + wo, C_, C_);
    convT_bf16<<<nCC, TPB, 0, stream>>>(Wk + wo, WkT + wo, C_, C_);
    convT_bf16<<<nCC, TPB, 0, stream>>>(Wv + wo, WvT + wo, C_, C_);
    convT_bf16<<<nCC, TPB, 0, stream>>>(Wo + wo, WoT + wo, C_, C_);
    size_t fo = (size_t)i * C_ * FF_;
    int nCF = (C_ * FF_ + TPB - 1) / TPB;
    convT_bf16<<<nCF, TPB, 0, stream>>>(W1 + fo, W1T + fo, C_, FF_);
    convT_bf16<<<nCF, TPB, 0, stream>>>(W2 + fo, W2T + fo, FF_, C_);
  }
  convT_bf16<<<(C_ * V_ + TPB - 1) / TPB, TPB, 0, stream>>>(Wh, WhT, C_, V_);

  // --- embedding ---
  embed_kernel<<<BT_, TPB, 0, stream>>>(idx, tok, pos, x);

  dim3 gQKV(C_ / 128, BT_ / 128);     // 4 x 16
  dim3 gFF (FF_ / 128, BT_ / 128);    // 16 x 16
  dim3 gHead(V_ / 128, BT_ / 128);    // 250 x 16
  dim3 gAttn(T_ / 128, B_ * H_);      // 8 x 16

  for (int i = 0; i < L_; i++) {
    size_t wo = (size_t)i * C_ * C_;
    size_t fo = (size_t)i * C_ * FF_;
    // h = LN1(x)
    layernorm_bf16<<<BT_, TPB, 0, stream>>>(x, ln1w + i * C_, ln1b + i * C_, hb);
    // q,k,v = h @ W + b   (bf16 out)
    gemm_bf16<<<gQKV, TPB, 0, stream>>>(hb, WqT + wo, bq + i * C_, nullptr,
                                        nullptr, qb, BT_, C_, C_, 0);
    gemm_bf16<<<gQKV, TPB, 0, stream>>>(hb, WkT + wo, bk + i * C_, nullptr,
                                        nullptr, kb, BT_, C_, C_, 0);
    gemm_bf16<<<gQKV, TPB, 0, stream>>>(hb, WvT + wo, bv + i * C_, nullptr,
                                        nullptr, vb, BT_, C_, C_, 0);
    // y = causal softmax(q k^T / 8) v
    attn_flash<<<gAttn, TPB, 0, stream>>>(qb, kb, vb, yb);
    // x = x + y @ Wo + bo
    gemm_bf16<<<gQKV, TPB, 0, stream>>>(yb, WoT + wo, bo + i * C_, x,
                                        x, nullptr, BT_, C_, C_, 0);
    // h = LN2(x)
    layernorm_bf16<<<BT_, TPB, 0, stream>>>(x, ln2w + i * C_, ln2b + i * C_, hb);
    // g = gelu(h @ W1 + b1)  (bf16 out)
    gemm_bf16<<<gFF, TPB, 0, stream>>>(hb, W1T + fo, b1 + i * FF_, nullptr,
                                       nullptr, gb, BT_, FF_, C_, 1);
    // x = x + g @ W2 + b2
    gemm_bf16<<<gQKV, TPB, 0, stream>>>(gb, W2T + fo, b2 + i * C_, x,
                                        x, nullptr, BT_, C_, FF_, 0);
  }
  // final LN + head
  layernorm_bf16<<<BT_, TPB, 0, stream>>>(x, lnfw, lnfb, hb);
  gemm_bf16<<<gHead, TPB, 0, stream>>>(hb, WhT, nullptr, nullptr,
                                       logits, nullptr, BT_, V_, C_, 0);
}
